// ds_us_fn_36575941493117
// MI455X (gfx1250) — matrix operand order repaired and checked by reading, not compiled, not run on hardware
//
#include <hip/hip_runtime.h>

// out[b,c,o] = sum_v M[o,v] * x[b,c,v]
//   == GEMM: Out(8192 x 1723) = X(8192 x 6890) @ M^T
// MI455X strategy: fp32 -> bf16 once (138 MB working set fits the 192 MB L2),
// then tiled GEMM on v_wmma_f32_16x16x32_bf16 (fp32 accum). Staging uses the
// CDNA5 async global->LDS path (ASYNCcnt) when the builtins exist, otherwise
// falls back to global_load_b128 + ds_store_b128.
// Workspace: Xb = 8192*6912 bf16 (113.2 MB) + Mb = 1792*6912 bf16 (24.8 MB)
// -> needs ws_size >= 138,018,816 bytes.

typedef __attribute__((ext_vector_type(16))) __bf16 v16bf;
typedef __attribute__((ext_vector_type(8)))  float  v8f;
typedef __attribute__((__vector_size__(4 * sizeof(int)))) int v4i;  // builtin's pointee type

#define NB   32
#define NC   256
#define VIN  6890
#define VOUT 1723
#define ROWS (NB * NC)   // 8192
#define KPAD 6912        // VIN padded to multiple of 32 (216 K-steps)
#define NPAD 1792        // VOUT padded to multiple of 128 (14 N-tiles)
#define BM   128
#define BN   128
#define BK   32
#define LDSK 40          // BK + 8 bf16 pad -> 80B row stride in LDS

#if defined(__has_builtin)
#if __has_builtin(__builtin_amdgcn_global_load_async_to_lds_b128) && \
    __has_builtin(__builtin_amdgcn_s_wait_asynccnt)
#define USE_ASYNC_LDS 1
#endif
#endif

static __device__ __forceinline__ unsigned short f2bf(float f) {
    unsigned int u = __float_as_uint(f);
    u += 0x7FFFu + ((u >> 16) & 1u);   // round-to-nearest-even
    return (unsigned short)(u >> 16);
}

// ---- fp32 -> bf16 conversion with zero padding -------------------------------
__global__ void cvt_x_kernel(const float* __restrict__ x, unsigned short* __restrict__ Xb) {
    size_t i = ((size_t)blockIdx.x * blockDim.x + threadIdx.x) * 8; // 8 elems/thread
    int row = (int)(i / KPAD);
    int k   = (int)(i % KPAD);        // multiple of 8; KPAD % 8 == 0 -> same row
    union { unsigned short s[8]; uint4 u; } v;
#pragma unroll
    for (int j = 0; j < 8; ++j) {
        int kk = k + j;
        v.s[j] = (kk < VIN) ? f2bf(x[(size_t)row * VIN + kk]) : (unsigned short)0;
    }
    *(uint4*)(Xb + i) = v.u;
}

__global__ void cvt_m_kernel(const float* __restrict__ M, unsigned short* __restrict__ Mb) {
    size_t i = ((size_t)blockIdx.x * blockDim.x + threadIdx.x) * 8;
    int row = (int)(i / KPAD);
    int k   = (int)(i % KPAD);
    union { unsigned short s[8]; uint4 u; } v;
#pragma unroll
    for (int j = 0; j < 8; ++j) {
        int kk = k + j;
        v.s[j] = (row < VOUT && kk < VIN) ? f2bf(M[(size_t)row * VIN + kk]) : (unsigned short)0;
    }
    *(uint4*)(Mb + i) = v.u;
}

// ---- WMMA GEMM ---------------------------------------------------------------
__global__ void __launch_bounds__(256, 2)
gemm_bf16_wmma_kernel(const unsigned short* __restrict__ Xb,
                      const unsigned short* __restrict__ Mb,
                      float* __restrict__ out)
{
    __shared__ __align__(16) unsigned short sA[2][BM * LDSK];
    __shared__ __align__(16) unsigned short sB[2][BN * LDSK];

    const int tid   = threadIdx.x;
    const int lane  = tid & 31;
    const int w     = tid >> 5;       // 8 waves
    const int waveM = w & 3;          // 4 waves along M (32 rows each)
    const int waveN = w >> 2;         // 2 waves along N (64 cols each)
    const int hi    = lane >> 4;      // lane half (K-group select)
    const int mr    = lane & 15;

    const int rowA0 = blockIdx.y * BM;   // X row base
    const int rowB0 = blockIdx.x * BN;   // M row base == output col base

    // Staging map: tile = 128 rows x 32 bf16 = 512 sixteen-byte segments,
    // 256 threads handle 2 segments each.
    const int s0 = tid, s1 = tid + 256;

    auto gA = [&](int s, int kb) {
        return (const uint4*)(Xb + (size_t)(rowA0 + (s >> 2)) * KPAD + kb + (s & 3) * 8);
    };
    auto gB = [&](int s, int kb) {
        return (const uint4*)(Mb + (size_t)(rowB0 + (s >> 2)) * KPAD + kb + (s & 3) * 8);
    };
    auto lP = [&](unsigned short* base, int s) {
        return (uint4*)(base + (s >> 2) * LDSK + (s & 3) * 8);
    };

#ifdef USE_ASYNC_LDS
    auto stage_async = [&](int buf, int kb) {
        __builtin_amdgcn_global_load_async_to_lds_b128((v4i*)gA(s0, kb), (v4i*)lP(sA[buf], s0), 0, 0);
        __builtin_amdgcn_global_load_async_to_lds_b128((v4i*)gA(s1, kb), (v4i*)lP(sA[buf], s1), 0, 0);
        __builtin_amdgcn_global_load_async_to_lds_b128((v4i*)gB(s0, kb), (v4i*)lP(sB[buf], s0), 0, 0);
        __builtin_amdgcn_global_load_async_to_lds_b128((v4i*)gB(s1, kb), (v4i*)lP(sB[buf], s1), 0, 0);
    };
#endif

    v8f acc[2][4];
#pragma unroll
    for (int mt = 0; mt < 2; ++mt)
#pragma unroll
        for (int nt = 0; nt < 4; ++nt)
#pragma unroll
            for (int r = 0; r < 8; ++r)
                acc[mt][nt][r] = 0.0f;

    // Prologue: stage K-tile 0 into buffer 0.
#ifdef USE_ASYNC_LDS
    stage_async(0, 0);
    __builtin_amdgcn_s_wait_asynccnt(0);
#else
    {
        uint4 a0 = *gA(s0, 0), a1 = *gA(s1, 0);
        uint4 b0 = *gB(s0, 0), b1 = *gB(s1, 0);
        *lP(sA[0], s0) = a0;  *lP(sA[0], s1) = a1;
        *lP(sB[0], s0) = b0;  *lP(sB[0], s1) = b1;
    }
#endif
    __syncthreads();

    const int NK = KPAD / BK;  // 216
#pragma unroll 2
    for (int it = 0; it < NK; ++it) {
        const int  buf  = it & 1;
        const bool more = (it + 1) < NK;
        const int  kbn  = (it + 1) * BK;

        // Software pipeline: next tile's staging first, overlapped with WMMA.
#ifdef USE_ASYNC_LDS
        if (more) stage_async(buf ^ 1, kbn);
#else
        uint4 a0, a1, b0, b1;
        if (more) {
            a0 = *gA(s0, kbn);  a1 = *gA(s1, kbn);
            b0 = *gB(s0, kbn);  b1 = *gB(s1, kbn);
        }
#endif

        // Load fragments from LDS per CDNA5 16-bit layouts.
        union { uint4 u[2]; v16bf v; } fa[2], fb[4];
#pragma unroll
        for (int mt = 0; mt < 2; ++mt) {
            // A 16x32: lane half hi holds K chunks {hi*8..hi*8+7, 16+hi*8..23+hi*8}
            const unsigned short* p = &sA[buf][(waveM * 32 + mt * 16 + mr) * LDSK];
            fa[mt].u[0] = *(const uint4*)(p + hi * 8);
            fa[mt].u[1] = *(const uint4*)(p + hi * 8 + 16);
        }
#pragma unroll
        for (int nt = 0; nt < 4; ++nt) {
            // B 32x16: column = lane%16 (row of M), K chunks {hi*8..hi*8+7, 16+hi*8..23+hi*8}
            const unsigned short* p = &sB[buf][(waveN * 64 + nt * 16 + mr) * LDSK];
            fb[nt].u[0] = *(const uint4*)(p + hi * 8);
            fb[nt].u[1] = *(const uint4*)(p + hi * 8 + 16);
        }

#pragma unroll
        for (int mt = 0; mt < 2; ++mt)
#pragma unroll
            for (int nt = 0; nt < 4; ++nt)
                acc[mt][nt] = __builtin_amdgcn_wmma_f32_16x16x32_bf16(
                    /*neg_a=*/false, fa[mt].v, /*neg_b=*/false, fb[nt].v,
                    /*c_mod=*/(short)0, acc[mt][nt],
                    /*reuse_a=*/false, /*reuse_b=*/false);

        if (more) {
#ifdef USE_ASYNC_LDS
            __builtin_amdgcn_s_wait_asynccnt(0);
#else
            *lP(sA[buf ^ 1], s0) = a0;  *lP(sA[buf ^ 1], s1) = a1;
            *lP(sB[buf ^ 1], s0) = b0;  *lP(sB[buf ^ 1], s1) = b1;
#endif
            __syncthreads();
        }
    }

    // Epilogue: D layout — row = vgpr + 8*hi, col = lane%16.
    const int gr = rowA0 + waveM * 32;
    const int gc = rowB0 + waveN * 64;
#pragma unroll
    for (int mt = 0; mt < 2; ++mt)
#pragma unroll
        for (int nt = 0; nt < 4; ++nt) {
            const int col = gc + nt * 16 + mr;
            if (col < VOUT) {
                const int rbase = gr + mt * 16 + hi * 8;
#pragma unroll
                for (int r = 0; r < 8; ++r)
                    out[(size_t)(rbase + r) * VOUT + col] = acc[mt][nt][r];
            }
        }
}

// ---- launcher ----------------------------------------------------------------
extern "C" void kernel_launch(void* const* d_in, const int* in_sizes, int n_in,
                              void* d_out, int out_size, void* d_ws, size_t ws_size,
                              hipStream_t stream) {
    const float* x = (const float*)d_in[0];   // (32, 256, 6890) fp32
    const float* M = (const float*)d_in[1];   // (1723, 6890) fp32
    float* out = (float*)d_out;               // (32, 256, 1723) fp32

    unsigned short* Xb = (unsigned short*)d_ws;               // 8192 x 6912 bf16
    unsigned short* Mb = Xb + (size_t)ROWS * KPAD;            // 1792 x 6912 bf16

    // 8192*6912/8 = 7,077,888 threads -> 27648 blocks of 256
    cvt_x_kernel<<<(int)(((size_t)ROWS * KPAD / 8) / 256), 256, 0, stream>>>(x, Xb);
    // 1792*6912/8 = 1,548,288 threads -> 6048 blocks of 256
    cvt_m_kernel<<<(int)(((size_t)NPAD * KPAD / 8) / 256), 256, 0, stream>>>(M, Mb);

    dim3 grid(NPAD / BN, ROWS / BM);   // (14, 64)
    gemm_bf16_wmma_kernel<<<grid, 256, 0, stream>>>(Xb, Mb, out);
}
